// DynamicsShaper_47356309406008
// MI455X (gfx1250) — matrix operand order repaired and checked by reading, not compiled, not run on hardware
//
#include <hip/hip_runtime.h>
#include <cstdint>

// ---------------- problem constants ----------------
#define B_      8
#define T_      160000
#define NSEG    512
#define CHUNK   128
#define NCHUNK  (T_ / CHUNK)                     // 1250 chunks per row
#define GROUP   16
#define NGROUP  ((NCHUNK + GROUP - 1) / GROUP)   // 79 groups per row
#define TILES   ((NCHUNK + 31) / 32)             // 40 tiles of 32 chunks per row

#define GAIN_MIN 0.1f
#define GAIN_MAX 2.0f
// constant-folded logs: ln(2*pi*20/16000), ln(pi), ln(0.0707), ln(2.0)
#define LOG_MIN_W (-4.8467345f)
#define LOG_MAX_W ( 1.1447299f)
#define LOG_MIN_Q (-2.6493099f)
#define LOG_MAX_Q ( 0.69314718f)

typedef __attribute__((ext_vector_type(16))) _Float16 v16h;
typedef __attribute__((ext_vector_type(8)))  float    v8f;

__device__ __forceinline__ float sigmoidf_(float z) {
    return 1.0f / (1.0f + __expf(-z));
}

// ---------------------------------------------------------------------------
// K1: deterministic bucket sums. segment_ids sorted along T => runs == value
// buckets. One thread per (row,bucket): two binary searches + sequential sum.
// acc layout: [B][NSEG][4] = {sum0,sum1,sum2,count}
// ---------------------------------------------------------------------------
__global__ void k_seg_sums(const int* __restrict__ seg,
                           const float* __restrict__ logits,
                           float* __restrict__ acc) {
    int idx = blockIdx.x * blockDim.x + threadIdx.x;
    if (idx >= B_ * NSEG) return;
    int b = idx / NSEG, s = idx % NSEG;
    const int* sr = seg + b * T_;

    int a = 0, e = T_;                 // lower_bound(s)
    while (a < e) { int m = (a + e) >> 1; if (sr[m] < s) a = m + 1; else e = m; }
    int lo = a;
    e = T_;                            // upper_bound(s)
    while (a < e) { int m = (a + e) >> 1; if (sr[m] <= s) a = m + 1; else e = m; }
    int hi = a;

    float s0 = 0.f, s1 = 0.f, s2 = 0.f;
    const float* lr = logits + (size_t)b * T_ * 3;
    for (int t = lo; t < hi; ++t) {
        __builtin_prefetch(lr + (t + 32) * 3, 0, 0);
        s0 += lr[t * 3 + 0];
        s1 += lr[t * 3 + 1];
        s2 += lr[t * 3 + 2];
    }
    float* o = acc + (size_t)idx * 4;
    o[0] = s0; o[1] = s1; o[2] = s2; o[3] = (float)(hi - lo);
}

// ---------------------------------------------------------------------------
// K1b: per-bucket coefficients. All transcendentals happen here (4096 threads)
// instead of per-sample (1.28M samples): coefficients are run-constant.
// ---------------------------------------------------------------------------
__global__ void k_bucket_coeffs(const float* __restrict__ acc,
                                float4* __restrict__ bC4,
                                float* __restrict__ bGain) {
    int idx = blockIdx.x * blockDim.x + threadIdx.x;
    if (idx >= B_ * NSEG) return;
    const float* a = acc + (size_t)idx * 4;
    float inv_cnt = 1.0f / fmaxf(a[3], 1.0f);
    float m0 = a[0] * inv_cnt, m1 = a[1] * inv_cnt, m2 = a[2] * inv_cnt;

    float gain  = GAIN_MIN + (GAIN_MAX - GAIN_MIN) * sigmoidf_(m0);
    float w = __expf(LOG_MIN_W + sigmoidf_(m1) * (LOG_MAX_W - LOG_MIN_W));
    float q = __expf(LOG_MIN_Q + sigmoidf_(m2) * (LOG_MAX_Q - LOG_MIN_Q));

    float cw = __cosf(w);
    float alpha = __sinf(w) / (2.0f * q);
    float inv_a0 = 1.0f / (1.0f + alpha);
    float b0 = 0.5f * (1.0f - cw) * inv_a0;
    float b1 = (1.0f - cw) * inv_a0;
    float a1 = (-2.0f * cw) * inv_a0;
    float a2 = (1.0f - alpha) * inv_a0;

    bC4[idx]   = make_float4(b0, b1, a1, a2);
    bGain[idx] = gain;
}

// ---------------------------------------------------------------------------
// K2: pure gather: per-sample coef/gain from bucket tables; x = noise*gain.
// ---------------------------------------------------------------------------
__global__ void k_gather(const float* __restrict__ noise,
                         const int* __restrict__ seg,
                         const float4* __restrict__ bC4,
                         const float* __restrict__ bGain,
                         float* __restrict__ xbuf,
                         float4* __restrict__ c4) {
    int idx = blockIdx.x * blockDim.x + threadIdx.x;
    if (idx >= B_ * T_) return;
    int b = idx / T_;
    int bs = b * NSEG + seg[idx];
    xbuf[idx] = noise[idx] * bGain[bs];
    c4[idx]   = bC4[bs];
}

// ---------------------------------------------------------------------------
// K3: per-chunk affine transform, one wave per 32-chunk tile. Stages x (16KB)
// and coef halves (32KB) into LDS via gfx1250 async global->LDS b128.
// state s_n = S_n s_{n-1} + (f_n,0), S_n = [[-a1,-a2],[1,0]].
// chunkH per chunk (8 floats): {m00,m01,v0, m10,m11,v1, 0,0}
// ---------------------------------------------------------------------------
__global__ void k_chunk_xform(const float* __restrict__ xbuf,
                              const float4* __restrict__ c4,
                              float* __restrict__ chunkH) {
    __shared__ float  sx[32 * CHUNK];   // 16 KB
    __shared__ float4 sc[32 * 64];      // 32 KB (half-tile of coefs)

    int bid = blockIdx.x;
    int b = bid / TILES, tile = bid % TILES;
    int lane = threadIdx.x;
    const float*  xr = xbuf + (size_t)b * T_;
    const float4* kr = c4   + (size_t)b * T_;
    int tileBase = tile * 32 * CHUNK;

    // async-stage x tile: 4096 floats, 32 x (32 lanes x 16B)
    for (int i = 0; i < 32; ++i) {
        int elem = (i * 32 + lane) * 4;
        int g = tileBase + elem;
        if (g > T_ - 4) g = T_ - 4;
        uint32_t ldsOff = (uint32_t)(uintptr_t)(&sx[elem]);
        const float* gp = xr + g;
        asm volatile("global_load_async_to_lds_b128 %0, %1, off"
                     :: "v"(ldsOff), "v"(gp) : "memory");
    }

    int c = tile * 32 + lane;
    bool valid = (c < NCHUNK);
    int base = c * CHUNK;
    float xm1 = 0.f, xm2 = 0.f;
    if (valid) {
        xm1 = (base >= 1) ? xr[base - 1] : 0.f;
        xm2 = (base >= 2) ? xr[base - 2] : 0.f;
    }
    float m00 = 1.f, m01 = 0.f, v0 = 0.f;
    float m10 = 0.f, m11 = 1.f, v1 = 0.f;

    for (int h = 0; h < 2; ++h) {
        // async-stage coef half: 2048 float4 samples
        for (int it = 0; it < 64; ++it) {
            int s  = it * 32 + lane;           // 0..2047
            int ci = s >> 6, il = s & 63;
            int gs = tileBase + ci * CHUNK + h * 64 + il;
            if (gs > T_ - 1) gs = T_ - 1;
            uint32_t ldsOff = (uint32_t)(uintptr_t)(&sc[s]);
            const float4* gp = kr + gs;
            asm volatile("global_load_async_to_lds_b128 %0, %1, off"
                         :: "v"(ldsOff), "v"(gp) : "memory");
        }
        asm volatile("s_wait_asynccnt 0x0" ::: "memory");
        __syncthreads();
        if (valid) {
            for (int il = 0; il < 64; ++il) {
                int i = h * 64 + il;
                float4 k = sc[lane * 64 + il];
                float  xn = sx[lane * CHUNK + i];
                float  f  = k.x * xn + k.y * xm1 + k.x * xm2;
                float nm00 = -k.z * m00 - k.w * m10;
                float nm01 = -k.z * m01 - k.w * m11;
                float nv0  = -k.z * v0  - k.w * v1 + f;
                m10 = m00; m11 = m01; v1 = v0;
                m00 = nm00; m01 = nm01; v0 = nv0;
                xm2 = xm1; xm1 = xn;
            }
        }
        __syncthreads();
    }
    if (valid) {
        float* o = chunkH + ((size_t)(b * NCHUNK + c)) * 8;
        o[0] = m00; o[1] = m01; o[2] = v0;
        o[3] = m10; o[4] = m11; o[5] = v1;
        o[6] = 0.f; o[7] = 0.f;
    }
}

// ---------------------------------------------------------------------------
// K4: WMMA group fold. One wave per (row, group of 16 chunks). Homogeneous 3x3
// transforms packed block-diagonally (5 per 16x16); each batch of up to 5 pairs is
// one v_wmma_f32_16x16x32_f16. Packing is branchless: each lane preloads the 3
// candidate values of its block row/column and fills fragments via selects.
// ---------------------------------------------------------------------------
__global__ void k_group_fold(const float* __restrict__ chunkH,
                             float* __restrict__ groupH) {
    __shared__ float sH[GROUP][8];
    __shared__ float sNx[8][8];
    __shared__ float sD[16][16];

    int gid = blockIdx.x;
    int b = gid / NGROUP, g = gid % NGROUP;
    int lane = threadIdx.x;

    // stage 16 chunk transforms (identity pad past row end)
    for (int i = lane; i < GROUP * 8; i += 32) {
        int k = i >> 3, j = i & 7;
        int c = g * GROUP + k;
        float idv = (j == 0 || j == 4) ? 1.f : 0.f;
        sH[k][j] = (c < NCHUNK) ? chunkH[((size_t)(b * NCHUNK + c)) * 8 + j] : idv;
    }
    __syncthreads();

    int M = lane & 15;
    int p = M / 3;                // block slot 0..4 (5 for M==15)
    int r = M - 3 * p;            // row (for A) == column (for B) within block
    int kbase = (lane < 16) ? 0 : 8;

    int cnt = GROUP;
    while (cnt > 1) {
        int npairs = cnt >> 1;
        for (int pbase = 0; pbase < npairs; pbase += 5) {
            __syncthreads();
            int  pi  = pbase + p;
            bool act = (M < 15) && (p < 5) && (pi < npairs);
            int  pcA = act ? (2 * pi + 1) : 0;   // H_later
            int  pcB = act ? (2 * pi)     : 0;   // H_earlier
            int  rs  = (r < 2) ? r : 0;

            // A row values: H2[r][0..2]; bottom row = (0,0,1)
            float l0 = sH[pcA][rs * 3 + 0];
            float l1 = sH[pcA][rs * 3 + 1];
            float l2 = sH[pcA][rs * 3 + 2];
            float a0_ = act ? (r < 2 ? l0 : 0.f) : 0.f;
            float a1_ = act ? (r < 2 ? l1 : 0.f) : 0.f;
            float a2_ = act ? (r < 2 ? l2 : 1.f) : 0.f;

            // B column values: H1[0..2][r]; row2 entry = (r==2)
            float g0 = sH[pcB][0 + r];
            float g1 = sH[pcB][3 + r];
            float b0_ = act ? g0 : 0.f;
            float b1_ = act ? g1 : 0.f;
            float b2_ = (act && r == 2) ? 1.f : 0.f;

            // ---- A fragment (16x32 f16): lanes0-15 M=lane, V0..3->K0..7,
            //      V4..7->K16..23; lanes16-31 same M, K += 8.
            v16h afrag;
            #pragma unroll
            for (int j = 0; j < 16; ++j) {
                int jv = j >> 1, half = j & 1;
                int K = ((jv < 4) ? (2 * jv) : (16 + 2 * (jv - 4))) + half + kbase;
                int cc = K - 3 * p;
                float v = (cc == 0) ? a0_ : (cc == 1) ? a1_ : (cc == 2) ? a2_ : 0.f;
                afrag[j] = (_Float16)v;
            }
            // ---- B fragment (32x16 f16): lanes0-15 N=lane, j0..7->K0..7,
            //      j8..15->K16..23; lanes16-31 same N, K += 8.
            v16h bfrag;
            #pragma unroll
            for (int j = 0; j < 16; ++j) {
                int K = ((j < 8) ? j : (8 + j)) + kbase;
                int rr = K - 3 * p;
                float v = (rr == 0) ? b0_ : (rr == 1) ? b1_ : (rr == 2) ? b2_ : 0.f;
                bfrag[j] = (_Float16)v;
            }

            v8f cacc = {};
            cacc = __builtin_amdgcn_wmma_f32_16x16x32_f16(
                false, afrag, false, bfrag, (short)0, cacc, false, false);

            // D layout: lanes0-15 N=lane, VGPR rr->M=rr; lanes16-31 -> M=rr+8
            #pragma unroll
            for (int rr = 0; rr < 8; ++rr) {
                int row = rr + ((lane < 16) ? 0 : 8);
                sD[row][M] = cacc[rr];
            }
            __syncthreads();
            // extract composed 3x3 blocks -> sNx
            if (lane < 30) {
                int pp = lane / 6, e2 = lane % 6;
                int pi2 = pbase + pp;
                if (pi2 < npairs) {
                    int rr = e2 / 3, cc = e2 % 3;
                    sNx[pi2][e2] = sD[3 * pp + rr][3 * pp + cc];
                }
            }
            __syncthreads();
        }
        // promote composed transforms (stride-8: shift/mask indexing)
        for (int i = lane; i < npairs * 8; i += 32)
            sH[i >> 3][i & 7] = sNx[i >> 3][i & 7];
        __syncthreads();
        cnt = npairs;
    }
    if (lane < 6)
        groupH[(size_t)gid * 8 + lane] = sH[0][lane];
}

// ---------------------------------------------------------------------------
// K5: tiny serial scan over 79 group transforms per row -> entry state/group.
// ---------------------------------------------------------------------------
__global__ void k_group_scan(const float* __restrict__ groupH,
                             float* __restrict__ groupSt) {
    int b = threadIdx.x;
    if (b >= B_) return;
    float y1 = 0.f, y2 = 0.f;
    for (int g = 0; g < NGROUP; ++g) {
        size_t gi = (size_t)(b * NGROUP + g);
        groupSt[gi * 2 + 0] = y1;
        groupSt[gi * 2 + 1] = y2;
        const float* H = groupH + gi * 8;
        float ny1 = H[0] * y1 + H[1] * y2 + H[2];
        float ny2 = H[3] * y1 + H[4] * y2 + H[5];
        y1 = ny1; y2 = ny2;
    }
}

// ---------------------------------------------------------------------------
// K6: within each group, walk 16 chunk transforms to get entry state/chunk.
// ---------------------------------------------------------------------------
__global__ void k_chunk_states(const float* __restrict__ chunkH,
                               const float* __restrict__ groupSt,
                               float* __restrict__ chunkSt) {
    int gid = blockIdx.x * blockDim.x + threadIdx.x;
    if (gid >= B_ * NGROUP) return;
    int b = gid / NGROUP, g = gid % NGROUP;
    float y1 = groupSt[(size_t)gid * 2 + 0];
    float y2 = groupSt[(size_t)gid * 2 + 1];
    for (int k = 0; k < GROUP; ++k) {
        int c = g * GROUP + k;
        if (c >= NCHUNK) break;
        size_t ci = (size_t)(b * NCHUNK + c);
        chunkSt[ci * 2 + 0] = y1;
        chunkSt[ci * 2 + 1] = y2;
        const float* H = chunkH + ci * 8;
        float ny1 = H[0] * y1 + H[1] * y2 + H[2];
        float ny2 = H[3] * y1 + H[4] * y2 + H[5];
        y1 = ny1; y2 = ny2;
    }
}

// ---------------------------------------------------------------------------
// K7: replay. One wave per 32-chunk tile; x + coef halves async-staged to LDS
// exactly like K3; each lane replays its 128-sample chunk from its entry state.
// ---------------------------------------------------------------------------
__global__ void k_replay(const float* __restrict__ xbuf,
                         const float4* __restrict__ c4,
                         const float* __restrict__ chunkSt,
                         float* __restrict__ out) {
    __shared__ float  sx[32 * CHUNK];   // 16 KB
    __shared__ float4 sc[32 * 64];      // 32 KB

    int bid = blockIdx.x;
    int b = bid / TILES, tile = bid % TILES;
    int lane = threadIdx.x;
    const float*  xr = xbuf + (size_t)b * T_;
    const float4* kr = c4   + (size_t)b * T_;
    int tileBase = tile * 32 * CHUNK;

    for (int i = 0; i < 32; ++i) {
        int elem = (i * 32 + lane) * 4;
        int g = tileBase + elem;
        if (g > T_ - 4) g = T_ - 4;
        uint32_t ldsOff = (uint32_t)(uintptr_t)(&sx[elem]);
        const float* gp = xr + g;
        asm volatile("global_load_async_to_lds_b128 %0, %1, off"
                     :: "v"(ldsOff), "v"(gp) : "memory");
    }

    int c = tile * 32 + lane;
    bool valid = (c < NCHUNK);
    int base = c * CHUNK;
    float y1 = 0.f, y2 = 0.f, xm1 = 0.f, xm2 = 0.f;
    if (valid) {
        size_t ci = (size_t)(b * NCHUNK + c);
        y1 = chunkSt[ci * 2 + 0];
        y2 = chunkSt[ci * 2 + 1];
        xm1 = (base >= 1) ? xr[base - 1] : 0.f;
        xm2 = (base >= 2) ? xr[base - 2] : 0.f;
    }
    float* orow = out + (size_t)b * T_;

    for (int h = 0; h < 2; ++h) {
        for (int it = 0; it < 64; ++it) {
            int s  = it * 32 + lane;
            int ci = s >> 6, il = s & 63;
            int gs = tileBase + ci * CHUNK + h * 64 + il;
            if (gs > T_ - 1) gs = T_ - 1;
            uint32_t ldsOff = (uint32_t)(uintptr_t)(&sc[s]);
            const float4* gp = kr + gs;
            asm volatile("global_load_async_to_lds_b128 %0, %1, off"
                         :: "v"(ldsOff), "v"(gp) : "memory");
        }
        asm volatile("s_wait_asynccnt 0x0" ::: "memory");
        __syncthreads();
        if (valid) {
            for (int il = 0; il < 64; ++il) {
                int i = h * 64 + il;
                int n = base + i;
                float4 k  = sc[lane * 64 + il];
                float  xn = sx[lane * CHUNK + i];
                float  y  = k.x * xn + k.y * xm1 + k.x * xm2 - k.z * y1 - k.w * y2;
                orow[n] = y;
                y2 = y1; y1 = y;
                xm2 = xm1; xm1 = xn;
            }
        }
        __syncthreads();
    }
}

// ---------------------------------------------------------------------------
extern "C" void kernel_launch(void* const* d_in, const int* in_sizes, int n_in,
                              void* d_out, int out_size, void* d_ws, size_t ws_size,
                              hipStream_t stream) {
    const float* noise  = (const float*)d_in[0];   // [8,160000]
    const int*   seg    = (const int*)  d_in[1];   // [8,160000]
    const float* logits = (const float*)d_in[2];   // [8,160000,3]
    float*       out    = (float*)d_out;           // [8,160000]

    // workspace carving (floats); float4 regions 16B aligned by construction
    float*  ws      = (float*)d_ws;
    float*  acc     = ws;                                   // B*NSEG*4
    float4* bC4     = (float4*)(acc + B_ * NSEG * 4);       // B*NSEG float4
    float*  bGain   = (float*)(bC4 + B_ * NSEG);            // B*NSEG
    float*  xbuf    = bGain + B_ * NSEG;                    // B*T
    float4* c4      = (float4*)(xbuf + (size_t)B_ * T_);    // B*T float4
    float*  chunkH  = (float*)(c4 + (size_t)B_ * T_);       // B*NCHUNK*8
    float*  groupH  = chunkH + (size_t)B_ * NCHUNK * 8;     // B*NGROUP*8
    float*  groupSt = groupH + (size_t)B_ * NGROUP * 8;     // B*NGROUP*2
    float*  chunkSt = groupSt + (size_t)B_ * NGROUP * 2;    // B*NCHUNK*2

    k_seg_sums<<<(B_ * NSEG + 255) / 256, 256, 0, stream>>>(seg, logits, acc);
    k_bucket_coeffs<<<(B_ * NSEG + 255) / 256, 256, 0, stream>>>(acc, bC4, bGain);
    k_gather<<<(B_ * T_ + 255) / 256, 256, 0, stream>>>(noise, seg, bC4, bGain, xbuf, c4);
    k_chunk_xform<<<B_ * TILES, 32, 0, stream>>>(xbuf, c4, chunkH);
    k_group_fold<<<B_ * NGROUP, 32, 0, stream>>>(chunkH, groupH);
    k_group_scan<<<1, 32, 0, stream>>>(groupH, groupSt);
    k_chunk_states<<<(B_ * NGROUP + 255) / 256, 256, 0, stream>>>(chunkH, groupSt, chunkSt);
    k_replay<<<B_ * TILES, 32, 0, stream>>>(xbuf, c4, chunkSt, out);

    (void)in_sizes; (void)n_in; (void)out_size; (void)ws_size;
}
